// Actor_55980603736134
// MI455X (gfx1250) — compile-verified, matrix operand order checked by reading, not run on hardware
//
#include <hip/hip_runtime.h>
#include <hip/hip_bf16.h>

typedef __bf16 v16bf __attribute__((ext_vector_type(16)));
typedef float  v8f   __attribute__((ext_vector_type(8)));

#define NROW_TILE 16
#define CIN 128
#define DIN 141
#define KPAD 160           // 5 k-steps of 32
#define HID 512
#define JNT 5

__device__ __forceinline__ float selu_f(float x) {
    const float sc = 1.0507009873554805f;
    const float al = 1.6732632423543772f;
    return x > 0.f ? sc * x : sc * al * (expf(x) - 1.f);
}

// ---- repack W1 (141x512 f32) -> bf16 B-fragments, K padded to 160 ----
// layout: frag(kt 0..4, nt 0..31): lane l holds 8 dwords (16 bf16), contiguous 32B.
// lane l: N = nt*16 + (l&15), half = l>>4; vgpr v: K = kt*32 + kbase(v) + 8*half, pair {K,K+1}
__global__ void prep_w1(const float* __restrict__ W1, __bf16* __restrict__ w1p) {
    int idx = blockIdx.x * 256 + threadIdx.x;       // 5*32*32*8 = 40960
    if (idx >= 5 * 32 * 32 * 8) return;
    int v    = idx & 7;
    int lane = (idx >> 3) & 31;
    int nt   = (idx >> 8) & 31;
    int kt   = idx >> 13;
    int n    = nt * 16 + (lane & 15);
    int half = lane >> 4;
    int kk   = (v < 4 ? 2 * v : 16 + 2 * (v - 4)) + 8 * half;
    int K0   = kt * 32 + kk;
    float a = (K0     < DIN) ? W1[(size_t)K0 * HID + n]       : 0.f;
    float b = (K0 + 1 < DIN) ? W1[(size_t)(K0 + 1) * HID + n] : 0.f;
    w1p[(size_t)idx * 2 + 0] = (__bf16)a;
    w1p[(size_t)idx * 2 + 1] = (__bf16)b;
}

__global__ void mask_zero(unsigned char* __restrict__ mask, int n) {
    int i = blockIdx.x * 256 + threadIdx.x;
    if (i < n) mask[i] = 0;
}
__global__ void mask_set(const int* __restrict__ coords, unsigned char* __restrict__ mask, int n) {
    int i = blockIdx.x * 256 + threadIdx.x;
    if (i < n) mask[coords[i]] = 1;
}

// ---- fused: mask-gate + LN1 + SELU + concat + GEMM1(bf16 WMMA) + LN2 + SELU + GEMM2 + tanh ----
__global__ __launch_bounds__(256) void actor_main(
    const float* __restrict__ feats, const unsigned char* __restrict__ mask,
    const float* __restrict__ jp, const float* __restrict__ jg, const float* __restrict__ wts,
    const float* __restrict__ ln1g, const float* __restrict__ ln1b,
    const float* __restrict__ b1,
    const float* __restrict__ ln2g, const float* __restrict__ ln2b,
    const float* __restrict__ W2, const float* __restrict__ b2,
    const __bf16* __restrict__ w1p, float* __restrict__ out)
{
    __shared__ __align__(32) __bf16 Afrag[5 * 32 * 16];   // 5 KB, ISA A-fragment layout
    __shared__ float hbuf[NROW_TILE][HID];                // 32 KB
    __shared__ float red[NROW_TILE][16];
    __shared__ float red2[NROW_TILE][16];
    __shared__ float mrow[NROW_TILE], rrow[NROW_TILE];
    __shared__ float p2[NROW_TILE][16][JNT];              // 5 KB

    const int t   = threadIdx.x;
    const int rr  = t >> 4;          // row in tile (0..15)
    const int c16 = t & 15;          // 16 threads per row
    const int grow = blockIdx.x * NROW_TILE + rr;

    // helper: write element (m,k) of the 16x160 bf16 A matrix into fragment layout
    auto awrite = [&](int mm, int k, float val) {
        int kt   = k >> 5, kk = k & 31;
        int half = (kk >> 3) & 1;
        int vv   = (kk < 16 ? 0 : 4) + ((kk & 7) >> 1);
        int pos  = kk & 1;
        int lane = half * 16 + mm;
        Afrag[((kt * 32 + lane) * 8 + vv) * 2 + pos] = (__bf16)val;
    };

    // ---- Phase A: masked load + LN1 stats ----
    float v[8];
    const float mgate = (float)mask[grow];
    float s = 0.f, sq = 0.f;
    const float* fr = feats + (size_t)grow * CIN + c16 * 8;
#pragma unroll
    for (int i = 0; i < 8; i++) { float x = fr[i] * mgate; v[i] = x; s += x; sq += x * x; }
    red[rr][c16] = s; red2[rr][c16] = sq;
    __syncthreads();
    if (t < NROW_TILE) {
        float ss = 0.f, qq = 0.f;
        for (int i = 0; i < 16; i++) { ss += red[t][i]; qq += red2[t][i]; }
        float mu  = ss * (1.f / CIN);
        float var = fmaxf(qq * (1.f / CIN) - mu * mu, 0.f);
        mrow[t] = mu; rrow[t] = rsqrtf(var + 1e-5f);
    }
    __syncthreads();

    // ---- Phase B: normalize + SELU + concat -> LDS A-fragments ----
    {
        float mu = mrow[rr], rs = rrow[rr];
#pragma unroll
        for (int i = 0; i < 8; i++) {
            int c = c16 * 8 + i;
            float xn = (v[i] - mu) * rs * ln1g[c] + ln1b[c];
            awrite(rr, 13 + c, selu_f(xn));
        }
    }
    for (int idx = t; idx < NROW_TILE * 13; idx += 256) {   // jnt_pos | jnt_goal | weights
        int mm = idx / 13, k = idx % 13;
        int gr = blockIdx.x * NROW_TILE + mm;
        float val = (k < 5)  ? jp[gr * JNT + k]
                  : (k < 10) ? jg[gr * JNT + (k - 5)]
                             : wts[gr * 3 + (k - 10)];
        awrite(mm, k, val);
    }
    for (int idx = t; idx < NROW_TILE * (KPAD - DIN); idx += 256) {  // zero pad K 141..159
        int mm = idx / (KPAD - DIN), k = DIN + idx % (KPAD - DIN);
        awrite(mm, k, 0.f);
    }
    __syncthreads();

    // ---- Phase C: GEMM1 via v_wmma_f32_16x16x32_bf16 ----
    const int wave = t >> 5, lane = t & 31;
    v8f acc[4] = {};
    const v16bf* Bp = reinterpret_cast<const v16bf*>(w1p);
#pragma unroll
    for (int kt = 0; kt < 5; kt++) {
        v16bf a = *reinterpret_cast<const v16bf*>(&Afrag[(kt * 32 + lane) * 16]);
#pragma unroll
        for (int j = 0; j < 4; j++) {
            int nt = wave * 4 + j;
            v16bf b = Bp[(size_t)(kt * 32 + nt) * 32 + lane];
            acc[j] = __builtin_amdgcn_wmma_f32_16x16x32_bf16(
                false, a, false, b, (short)0, acc[j], false, false);
        }
    }
    // scatter D tiles (+b1) to LDS: vgpr jj -> row jj + 8*(lane>>4), col nt*16 + (lane&15)
    {
        int rbase = (lane >> 4) * 8;
#pragma unroll
        for (int j = 0; j < 4; j++) {
            int col = (wave * 4 + j) * 16 + (lane & 15);
            float bb = b1[col];
#pragma unroll
            for (int jj = 0; jj < 8; jj++)
                hbuf[rbase + jj][col] = acc[j][jj] + bb;
        }
    }
    __syncthreads();

    // ---- Phase D: LN2 stats (512) ----
    float s2 = 0.f, q2 = 0.f;
#pragma unroll 4
    for (int i = 0; i < 32; i++) {
        float h = hbuf[rr][c16 * 32 + i];
        s2 += h; q2 += h * h;
    }
    red[rr][c16] = s2; red2[rr][c16] = q2;
    __syncthreads();
    if (t < NROW_TILE) {
        float ss = 0.f, qq = 0.f;
        for (int i = 0; i < 16; i++) { ss += red[t][i]; qq += red2[t][i]; }
        float mu  = ss * (1.f / HID);
        float var = fmaxf(qq * (1.f / HID) - mu * mu, 0.f);
        mrow[t] = mu; rrow[t] = rsqrtf(var + 1e-5f);
    }
    __syncthreads();

    // ---- Phase E: SELU + GEMM2 (512->5, VALU) ----
    {
        float mu = mrow[rr], rs = rrow[rr];
        float a0 = 0.f, a1 = 0.f, a2 = 0.f, a3 = 0.f, a4 = 0.f;
#pragma unroll 4
        for (int i = 0; i < 32; i++) {
            int n = c16 * 32 + i;
            float xn = selu_f((hbuf[rr][n] - mu) * rs * ln2g[n] + ln2b[n]);
            const float* w = W2 + (size_t)n * JNT;
            a0 += xn * w[0]; a1 += xn * w[1]; a2 += xn * w[2];
            a3 += xn * w[3]; a4 += xn * w[4];
        }
        p2[rr][c16][0] = a0; p2[rr][c16][1] = a1; p2[rr][c16][2] = a2;
        p2[rr][c16][3] = a3; p2[rr][c16][4] = a4;
    }
    __syncthreads();
    if (t < NROW_TILE * JNT) {
        int row = t / JNT, j = t % JNT;
        float ssum = b2[j];
        for (int i = 0; i < 16; i++) ssum += p2[row][i][j];
        out[(size_t)(blockIdx.x * NROW_TILE + row) * JNT + j] = tanhf(ssum) * 10.0f;
    }
}

extern "C" void kernel_launch(void* const* d_in, const int* in_sizes, int n_in,
                              void* d_out, int out_size, void* d_ws, size_t ws_size,
                              hipStream_t stream) {
    const float* feats = (const float*)d_in[0];
    const int*   coords= (const int*)  d_in[1];
    const float* jp    = (const float*)d_in[2];
    const float* jg    = (const float*)d_in[3];
    const float* wts   = (const float*)d_in[4];
    const float* ln1g  = (const float*)d_in[5];
    const float* ln1b  = (const float*)d_in[6];
    const float* W1    = (const float*)d_in[7];
    const float* b1    = (const float*)d_in[8];
    const float* ln2g  = (const float*)d_in[9];
    const float* ln2b  = (const float*)d_in[10];
    const float* W2    = (const float*)d_in[11];
    const float* b2    = (const float*)d_in[12];
    float* outp = (float*)d_out;

    const int n = in_sizes[0] / CIN;                    // 262144

    __bf16* w1p = (__bf16*)d_ws;                        // 160 KB bf16 fragments
    unsigned char* mask = (unsigned char*)d_ws + 163840; // N bytes

    prep_w1 <<<160, 256, 0, stream>>>(W1, w1p);
    mask_zero<<<(n + 255) / 256, 256, 0, stream>>>(mask, n);
    mask_set <<<(n + 255) / 256, 256, 0, stream>>>(coords, mask, n);
    actor_main<<<n / NROW_TILE, 256, 0, stream>>>(
        feats, mask, jp, jg, wts, ln1g, ln1b, b1, ln2g, ln2b, W2, b2, w1p, outp);
}
